// StackedPTBlock_47158740910872
// MI455X (gfx1250) — compile-verified, matrix operand order checked by reading, not run on hardware
//
#include <hip/hip_runtime.h>

// ---------------- problem constants (match reference setup) ----------------
#define BSC   2          // scenes
#define NPER  8192       // points per scene
#define NPTS  (BSC*NPER) // 16384 total points
#define KNN   16         // neighbors
#define HD    128        // hidden = in dim
#define LDP   136        // LDS pitch in halves (128 + 8 pad, keeps 16B align)
#define NKROW (NPTS*KNN) // 262144 (n,k) rows
#define LBLK  2          // stacked blocks
#define EPSBN 1e-5f

typedef __attribute__((ext_vector_type(16))) _Float16 v16h;
typedef __attribute__((ext_vector_type(8)))  _Float16 v8h;
typedef __attribute__((ext_vector_type(8)))  float    v8f;

// ---------------------------------------------------------------------------
// KNN: one thread per point, brute force over the scene, coords tiled in LDS,
// register-resident 16-deep insertion chain (ties keep lower index = top_k).
// ---------------------------------------------------------------------------
__global__ __launch_bounds__(256) void knn_kernel(const int* __restrict__ coords,
                                                  int* __restrict__ gidx,
                                                  float* __restrict__ rel)
{
  const int b = blockIdx.y;
  const int i = blockIdx.x * 256 + threadIdx.x;       // 0..NPER-1
  const int* cb = coords + (size_t)b * NPER * 3;
  const float xi = (float)cb[i*3+0];
  const float yi = (float)cb[i*3+1];
  const float zi = (float)cb[i*3+2];

  float bd[KNN]; int bi[KNN];
  #pragma unroll
  for (int k = 0; k < KNN; ++k) { bd[k] = 3.4e38f; bi[k] = 0; }

  __shared__ float cs[256*3];
  for (int jt = 0; jt < NPER; jt += 256) {
    __syncthreads();
    for (int t = threadIdx.x; t < 256*3; t += 256) cs[t] = (float)cb[jt*3 + t];
    __syncthreads();
    for (int jj = 0; jj < 256; ++jj) {
      const float dx = cs[jj*3+0] - xi;
      const float dy = cs[jj*3+1] - yi;
      const float dz = cs[jj*3+2] - zi;
      float cd = dx*dx + dy*dy + dz*dz;
      if (cd < bd[KNN-1]) {
        int ci = jt + jj;
        #pragma unroll
        for (int k = 0; k < KNN; ++k) {
          if (cd < bd[k]) { float td = bd[k]; int ti = bi[k];
                            bd[k] = cd; bi[k] = ci; cd = td; ci = ti; }
        }
      }
    }
  }

  const int n = b * NPER + i;
  #pragma unroll
  for (int k = 0; k < KNN; ++k) {
    const int j = bi[k];
    gidx[(size_t)n*KNN + k] = b * NPER + j;
    rel[((size_t)n*KNN + k)*3 + 0] = (float)cb[j*3+0] - xi;
    rel[((size_t)n*KNN + k)*3 + 1] = (float)cb[j*3+1] - yi;
    rel[((size_t)n*KNN + k)*3 + 2] = (float)cb[j*3+2] - zi;
  }
}

// ---------------------------------------------------------------------------
// WMMA GEMM: C[M x 128] = A[M x 128] @ B[128 x 128] (+bias), f16 operands,
// f32 accumulation.  Block = 256 threads = 8 waves; block owns a 64x128
// output slab (4 m-tiles), wave w owns column tile w.  B is staged
// TRANSPOSED in LDS (Bt[n][k], padded pitch) so every fragment -- A and B --
// is exactly two 16-byte ds_load_b128 per 16x16x32 step.  Each wave caches
// its 4 B fragments in registers and sweeps the 4 m-tiles (16 WMMAs/wave).
// Optional fused BN statistics (per-channel sum / sum^2), 2 atomics per lane.
// ---------------------------------------------------------------------------
__global__ __launch_bounds__(256) void gemm128_wmma(const float* __restrict__ A,
                                                    const float* __restrict__ B,
                                                    const float* __restrict__ bias,
                                                    float* __restrict__ C,
                                                    float* __restrict__ stats)
{
  __shared__ _Float16 Bt[HD*LDP];   // transposed weights, 34 KB
  __shared__ _Float16 As[64*LDP];   // 64-row activation slab, 17 KB
  const int tid = threadIdx.x;
  const size_t m0 = (size_t)blockIdx.x * 64;

  __builtin_prefetch(A + m0 * HD, 0, 3);          // global_prefetch

  // stage B transposed: read row-major coalesced, write Bt[n*LDP + k]
  for (int t = tid; t < HD*HD; t += 256) {
    const int k = t >> 7, n = t & 127;
    Bt[n*LDP + k] = (_Float16)B[t];
  }
  // stage A slab (64 x 128) with padded pitch
  for (int t = tid; t < 64*HD; t += 256) {
    const int r = t >> 7, c = t & 127;
    As[r*LDP + c] = (_Float16)A[m0*HD + (size_t)t];
  }
  __syncthreads();

  const int wave = tid >> 5;                 // column tile 0..7
  const int lane = tid & 31;
  const int nl   = lane & 15;                // row (A) / col (B) within tile
  const int klo  = (lane >> 4) << 3;         // 0 or 8 (ISA 16-bit layout)
  const int n    = wave * 16 + nl;           // output column

  // B fragments for all 4 k-steps, kept in registers
  v16h bf[4];
  #pragma unroll
  for (int s = 0; s < 4; ++s) {
    const v8h b0 = *(const v8h*)&Bt[n*LDP + s*32 + klo];
    const v8h b1 = *(const v8h*)&Bt[n*LDP + s*32 + 16 + klo];
    bf[s] = __builtin_shufflevector(b0, b1, 0,1,2,3,4,5,6,7,8,9,10,11,12,13,14,15);
  }

  const float bv = bias ? bias[n] : 0.0f;
  float s1 = 0.0f, s2 = 0.0f;

  #pragma unroll
  for (int mt = 0; mt < 4; ++mt) {
    v8f acc = {};
    #pragma unroll
    for (int s = 0; s < 4; ++s) {
      const v8h a0 = *(const v8h*)&As[(mt*16 + nl)*LDP + s*32 + klo];
      const v8h a1 = *(const v8h*)&As[(mt*16 + nl)*LDP + s*32 + 16 + klo];
      const v16h a = __builtin_shufflevector(a0, a1, 0,1,2,3,4,5,6,7,8,9,10,11,12,13,14,15);
      acc = __builtin_amdgcn_wmma_f32_16x16x32_f16(false, a, false, bf[s],
                                                   (short)0, acc, false, false);
    }
    #pragma unroll
    for (int r = 0; r < 8; ++r) {
      const size_t m = m0 + (size_t)(mt*16 + r + klo);  // lanes>=16 -> M+8
      const float v = acc[r] + bv;
      C[m * HD + n] = v;
      s1 += v; s2 += v * v;
    }
  }
  if (stats) { atomicAdd(&stats[n], s1); atomicAdd(&stats[HD + n], s2); }
}

// ---------------------------------------------------------------------------
// BN apply: y = (x-mean)*rsqrt(var+eps)*g + b  [+relu] [+residual]
// stats[ch] = sum, stats[128+ch] = sumsq, over `rows`.
// ---------------------------------------------------------------------------
__global__ __launch_bounds__(256) void bn_apply(const float* __restrict__ x,
                                                const float* __restrict__ stats,
                                                const float* __restrict__ g,
                                                const float* __restrict__ b,
                                                const float* __restrict__ res,
                                                float* __restrict__ y,
                                                float inv_rows, int do_relu)
{
  const size_t i  = (size_t)blockIdx.x * 256 + threadIdx.x;
  const int    ch = (int)(i & (HD - 1));
  const float mean = stats[ch] * inv_rows;
  const float var  = stats[HD + ch] * inv_rows - mean * mean;
  const float sc   = rsqrtf(var + EPSBN) * g[ch];
  const float sh   = b[ch] - mean * sc;
  float v = x[i] * sc + sh;
  if (do_relu) v = fmaxf(v, 0.0f);
  if (res) v += res[i];
  y[i] = v;
}

// ---------------------------------------------------------------------------
// delta MLP stage 1: out[row,ch] = rel[row,:3] . W[:,ch] + bias[ch], fused
// stats.  Block = 128 threads (one channel each), 64 rows per block.
// ---------------------------------------------------------------------------
__global__ __launch_bounds__(128) void pose_d1_kernel(const float* __restrict__ rel,
                                                      const float* __restrict__ W,
                                                      const float* __restrict__ bias,
                                                      float* __restrict__ out,
                                                      float* __restrict__ stats)
{
  __shared__ float rl[64*3];
  const int tid = threadIdx.x;
  const size_t r0 = (size_t)blockIdx.x * 64;
  for (int t = tid; t < 64*3; t += 128) rl[t] = rel[r0*3 + t];
  __syncthreads();
  const float w0 = W[0*HD + tid], w1 = W[1*HD + tid], w2 = W[2*HD + tid];
  const float bb = bias[tid];
  float s = 0.0f, s2 = 0.0f;
  for (int r = 0; r < 64; ++r) {
    const float v = rl[r*3+0]*w0 + rl[r*3+1]*w1 + rl[r*3+2]*w2 + bb;
    out[(r0 + r) * HD + tid] = v;
    s += v; s2 += v * v;
  }
  atomicAdd(&stats[tid], s); atomicAdd(&stats[HD + tid], s2);
}

// a_pre[n,k,ch] = q[n,ch] - kpsi[gidx[n,k],ch] + pose[n,k,ch]
__global__ __launch_bounds__(256) void compose_kernel(const float* __restrict__ q,
                                                      const float* __restrict__ kpsi,
                                                      const float* __restrict__ pose,
                                                      const int* __restrict__ gidx,
                                                      float* __restrict__ out)
{
  const size_t i   = (size_t)blockIdx.x * 256 + threadIdx.x;
  const size_t row = i >> 7;               // n*KNN + k
  const int    ch  = (int)(i & (HD - 1));
  const size_t n   = row >> 4;             // KNN = 16
  const int    gi  = gidx[row];
  out[i] = q[n * HD + ch] - kpsi[(size_t)gi * HD + ch] + pose[i];
}

// per-(n,ch) softmax over K neighbors; y = sum_k attn * (v[gidx]+pose)
__global__ __launch_bounds__(256) void softmax_agg_kernel(const float* __restrict__ a,
                                                          const float* __restrict__ valp,
                                                          const float* __restrict__ pose,
                                                          const int* __restrict__ gidx,
                                                          float* __restrict__ y)
{
  const size_t i  = (size_t)blockIdx.x * 256 + threadIdx.x;
  const size_t n  = i >> 7;
  const int    ch = (int)(i & (HD - 1));
  float vals[KNN];
  float mx = -3.4e38f;
  #pragma unroll
  for (int k = 0; k < KNN; ++k) {
    vals[k] = a[((n << 4) + k) * HD + ch];
    mx = fmaxf(mx, vals[k]);
  }
  float se = 0.0f;
  #pragma unroll
  for (int k = 0; k < KNN; ++k) { vals[k] = __expf(vals[k] - mx); se += vals[k]; }
  float acc = 0.0f;
  #pragma unroll
  for (int k = 0; k < KNN; ++k) {
    const size_t row = (n << 4) + k;
    const int gi = gidx[row];
    acc += vals[k] * (valp[(size_t)gi * HD + ch] + pose[row * HD + ch]);
  }
  y[i] = acc / se;
}

__global__ void zero_kernel(float* __restrict__ p, int n)
{
  const int i = blockIdx.x * 256 + threadIdx.x;
  if (i < n) p[i] = 0.0f;
}

// ---------------------------------------------------------------------------
extern "C" void kernel_launch(void* const* d_in, const int* in_sizes, int n_in,
                              void* d_out, int out_size, void* d_ws, size_t ws_size,
                              hipStream_t stream)
{
  (void)in_sizes; (void)n_in; (void)out_size; (void)ws_size;

  const int*   coords    = (const int*)  d_in[0];
  const float* feats     = (const float*)d_in[1];
  const float* W_top     = (const float*)d_in[2];
  const float* bn_top_g  = (const float*)d_in[3];
  const float* bn_top_b  = (const float*)d_in[4];
  const float* W_phi     = (const float*)d_in[5];
  const float* W_psi     = (const float*)d_in[6];
  const float* W_alpha   = (const float*)d_in[7];
  const float* W_d1      = (const float*)d_in[8];
  const float* b_d1      = (const float*)d_in[9];
  const float* bn_d1_g   = (const float*)d_in[10];
  const float* bn_d1_b   = (const float*)d_in[11];
  const float* W_d2      = (const float*)d_in[12];
  const float* b_d2      = (const float*)d_in[13];
  const float* bn_d2_g   = (const float*)d_in[14];
  const float* bn_d2_b   = (const float*)d_in[15];
  const float* W_g1      = (const float*)d_in[16];
  const float* b_g1      = (const float*)d_in[17];
  const float* bn_g1_g   = (const float*)d_in[18];
  const float* bn_g1_b   = (const float*)d_in[19];
  const float* W_g2      = (const float*)d_in[20];
  const float* b_g2      = (const float*)d_in[21];
  const float* bn_g2_g   = (const float*)d_in[22];
  const float* bn_g2_b   = (const float*)d_in[23];
  const float* W_down    = (const float*)d_in[24];
  const float* bn_down_g = (const float*)d_in[25];
  const float* bn_down_b = (const float*)d_in[26];

  float* f = (float*)d_out;                     // running features (N x 128)

  // ---- workspace carve-up (floats) ----
  float* ws = (float*)d_ws;
  size_t o = 0;
  int*   gidx  = (int*)(ws + o); o += (size_t)NKROW;        // N*K ints
  float* rel   = ws + o;         o += (size_t)NKROW * 3;
  float* stats = ws + o;         o += 256;
  float* hbuf  = ws + o;         o += (size_t)NPTS * HD;
  float* qbuf  = ws + o;         o += (size_t)NPTS * HD;
  float* kpsi  = ws + o;         o += (size_t)NPTS * HD;
  float* valp  = ws + o;         o += (size_t)NPTS * HD;
  float* t1    = ws + o;         o += (size_t)NPTS * HD;
  float* ybuf  = ws + o;         o += (size_t)NPTS * HD;
  float* pose  = ws + o;         o += (size_t)NKROW * HD;
  float* bufA  = ws + o;         o += (size_t)NKROW * HD;
  float* bufB  = ws + o;         o += (size_t)NKROW * HD;

  const unsigned gN   = NPTS  / 64;      // gemm blocks (64-row slabs)
  const unsigned gNK  = NKROW / 64;
  const unsigned eN   = (unsigned)((size_t)NPTS  * HD / 256);
  const unsigned eNK  = (unsigned)((size_t)NKROW * HD / 256);
  const float invN  = 1.0f / (float)NPTS;
  const float invNK = 1.0f / (float)NKROW;

  // KNN grouping + relative coords
  knn_kernel<<<dim3(NPER/256, BSC), 256, 0, stream>>>(coords, gidx, rel);

  // f <- feats
  hipMemcpyAsync(f, feats, (size_t)NPTS * HD * sizeof(float),
                 hipMemcpyDeviceToDevice, stream);

  for (int l = 0; l < LBLK; ++l) {
    const float* Wt  = W_top   + (size_t)l*HD*HD;
    const float* Wp  = W_phi   + (size_t)l*HD*HD;
    const float* Ws  = W_psi   + (size_t)l*HD*HD;
    const float* Wa  = W_alpha + (size_t)l*HD*HD;
    const float* Wd1 = W_d1    + (size_t)l*3*HD;
    const float* Wd2 = W_d2    + (size_t)l*HD*HD;
    const float* Wg1 = W_g1    + (size_t)l*HD*HD;
    const float* Wg2 = W_g2    + (size_t)l*HD*HD;
    const float* Wdn = W_down  + (size_t)l*HD*HD;

    // h = BN(f @ W_top)
    zero_kernel<<<1, 256, 0, stream>>>(stats, 256);
    gemm128_wmma<<<gN, 256, 0, stream>>>(f, Wt, nullptr, t1, stats);
    bn_apply<<<eN, 256, 0, stream>>>(t1, stats, bn_top_g + l*HD, bn_top_b + l*HD,
                                     nullptr, hbuf, invN, 0);

    // pose = BN(relu(BN(rel@W_d1 + b_d1)) @ W_d2 + b_d2)
    zero_kernel<<<1, 256, 0, stream>>>(stats, 256);
    pose_d1_kernel<<<NKROW/64, 128, 0, stream>>>(rel, Wd1, b_d1 + l*HD, bufA, stats);
    bn_apply<<<eNK, 256, 0, stream>>>(bufA, stats, bn_d1_g + l*HD, bn_d1_b + l*HD,
                                      nullptr, bufB, invNK, 1);
    zero_kernel<<<1, 256, 0, stream>>>(stats, 256);
    gemm128_wmma<<<gNK, 256, 0, stream>>>(bufB, Wd2, b_d2 + l*HD, bufA, stats);
    bn_apply<<<eNK, 256, 0, stream>>>(bufA, stats, bn_d2_g + l*HD, bn_d2_b + l*HD,
                                      nullptr, pose, invNK, 0);

    // q / psi / alpha projections (no BN)
    gemm128_wmma<<<gN, 256, 0, stream>>>(hbuf, Wp, nullptr, qbuf, nullptr);
    gemm128_wmma<<<gN, 256, 0, stream>>>(hbuf, Ws, nullptr, kpsi, nullptr);
    gemm128_wmma<<<gN, 256, 0, stream>>>(hbuf, Wa, nullptr, valp, nullptr);

    // a = q - psi[gidx] + pose ; gamma MLP
    compose_kernel<<<eNK, 256, 0, stream>>>(qbuf, kpsi, pose, gidx, bufA);
    zero_kernel<<<1, 256, 0, stream>>>(stats, 256);
    gemm128_wmma<<<gNK, 256, 0, stream>>>(bufA, Wg1, b_g1 + l*HD, bufB, stats);
    bn_apply<<<eNK, 256, 0, stream>>>(bufB, stats, bn_g1_g + l*HD, bn_g1_b + l*HD,
                                      nullptr, bufA, invNK, 1);
    zero_kernel<<<1, 256, 0, stream>>>(stats, 256);
    gemm128_wmma<<<gNK, 256, 0, stream>>>(bufA, Wg2, b_g2 + l*HD, bufB, stats);
    bn_apply<<<eNK, 256, 0, stream>>>(bufB, stats, bn_g2_g + l*HD, bn_g2_b + l*HD,
                                      nullptr, bufA, invNK, 0);

    // softmax over K + aggregation -> y ; f = BN(y @ W_down) + f
    softmax_agg_kernel<<<eN, 256, 0, stream>>>(bufA, valp, pose, gidx, ybuf);
    zero_kernel<<<1, 256, 0, stream>>>(stats, 256);
    gemm128_wmma<<<gN, 256, 0, stream>>>(ybuf, Wdn, nullptr, t1, stats);
    bn_apply<<<eN, 256, 0, stream>>>(t1, stats, bn_down_g + l*HD, bn_down_b + l*HD,
                                     f, f, invN, 0);
  }
}
